// SimpleNN_58755152609950
// MI455X (gfx1250) — matrix operand order repaired and checked by reading, not compiled, not run on hardware
//
#include <hip/hip_runtime.h>

typedef __attribute__((ext_vector_type(16))) _Float16 v16h;
typedef __attribute__((ext_vector_type(8)))  float    v8f;
typedef __attribute__((ext_vector_type(8)))  unsigned v8u;
typedef __attribute__((ext_vector_type(4)))  unsigned v4u;
typedef __attribute__((ext_vector_type(4)))  int      v4i;
typedef __attribute__((ext_vector_type(4)))  float    vf4;
typedef __attribute__((ext_vector_type(2)))  float    vf2;

typedef __attribute__((address_space(1))) v4i g_v4i;   // global int4
typedef __attribute__((address_space(3))) v4i l_v4i;   // LDS int4

#define DEQ_ITERS 20
#define D_IN  784
#define D_HID 64
#define D_OUT 10
#define KCH   25          // ceil(784/32) k-chunks for fc1
#define PH0   13          // chunks staged in LDS phase 0
#define PH1   12          // chunks staged in LDS phase 1
#define ROWS_PER_WAVE  16
#define ROWS_PER_BLOCK 128
#define FRAG_DW 256       // one A-fragment = 32 lanes x 8 dwords
#define W1_DW  (KCH * 4 * FRAG_DW)   // 25600 dwords = 100 KiB in d_ws

#if defined(__gfx1250__) && __has_builtin(__builtin_amdgcn_global_load_async_to_lds_b128)
#define USE_ASYNC_LDS 1
#else
#define USE_ASYNC_LDS 0
#endif

// ---------- helpers ----------
__device__ __forceinline__ unsigned pack2(float a, float b) {
    _Float16 ha = (_Float16)a, hb = (_Float16)b;
    unsigned short ua = __builtin_bit_cast(unsigned short, ha);
    unsigned short ub = __builtin_bit_cast(unsigned short, hb);
    return (unsigned)ua | ((unsigned)ub << 16);
}

__device__ __forceinline__ v16h mk(const unsigned u[8]) {
    v8u t;
#pragma unroll
    for (int j = 0; j < 8; ++j) t[j] = u[j];
    return __builtin_bit_cast(v16h, t);
}

// branch-free tanh: exact saturation, ~5 VALU ops (v_exp_f32 based)
__device__ __forceinline__ float fast_tanh(float x) {
    float e = __expf(2.f * x);
    return 1.f - 2.f / (e + 1.f);
}

// A-fragment of W^T (16x32 f16, tile mt of output features, k-chunk c),
// gathered from row-major W[k][n] (ld = row length, ncols = valid n).
__device__ __forceinline__ v16h load_wA(const float* __restrict__ W, int ld,
                                        int ncols, int c, int mt, int lane) {
    const int m  = mt * 16 + (lane & 15);
    const int hi = lane >> 4;
    unsigned u[8];
#pragma unroll
    for (int v = 0; v < 8; ++v) {
        int k = c * 32 + 2 * v + 16 * hi;
        float a = (m < ncols) ? W[k * ld + m]       : 0.f;
        float b = (m < ncols) ? W[(k + 1) * ld + m] : 0.f;
        u[v] = pack2(a, b);
    }
    return mk(u);
}

// Build B operand (32 features x 16 batch) for chunk covering tiles (2c, 2c+1)
// from per-lane packed D-tiles via a single lane^16 exchange.
__device__ __forceinline__ v16h bfrag(const unsigned Plo[4], const unsigned Phi[4], int hi) {
    unsigned u[8];
#pragma unroll
    for (int j = 0; j < 4; ++j) {
        unsigned snd = hi ? Plo[j] : Phi[j];
        unsigned rcv = (unsigned)__shfl_xor((int)snd, 16, 32);
        u[j]     = hi ? rcv    : Plo[j];
        u[4 + j] = hi ? Phi[j] : rcv;
    }
    return mk(u);
}

__device__ __forceinline__ v8f wmma_f16(v16h a, v16h b, v8f c) {
    return __builtin_amdgcn_wmma_f32_16x16x32_f16(false, a, false, b,
                                                  (short)0, c, false, false);
}

// ---------- pre-pass: build pre-swizzled f16 W1^T fragments in d_ws ----------
__global__ void __launch_bounds__(256)
w1_pack_kernel(const float* __restrict__ W1, unsigned* __restrict__ wsFrag) {
    int idx = blockIdx.x * 256 + threadIdx.x;
    if (idx >= W1_DW) return;
    int f   = idx >> 8;          // fragment id = c*4 + mt
    int rem = idx & 255;
    int v   = rem & 7;
    int lam = rem >> 3;
    int c   = f >> 2;
    int mt  = f & 3;
    int m   = mt * 16 + (lam & 15);
    int h2  = lam >> 4;
    int k   = c * 32 + 2 * v + 16 * h2;
    float a = (k     < D_IN) ? W1[k * D_HID + m]       : 0.f;
    float b = (k + 1 < D_IN) ? W1[(k + 1) * D_HID + m] : 0.f;
    wsFrag[idx] = pack2(a, b);
}

// ---------- main kernel ----------
__global__ void __launch_bounds__(256)
deq_wmma_kernel(const float* __restrict__ x,
                const float* __restrict__ b1,
                const float* __restrict__ Wx, const float* __restrict__ bx,
                const float* __restrict__ Wz, const float* __restrict__ bz,
                const float* __restrict__ W3, const float* __restrict__ b3,
                const unsigned* __restrict__ wsFrag,
                float* __restrict__ out, int B)
{
    __shared__ unsigned sFrag[PH0 * 4 * FRAG_DW];   // 53,248 B

    const int tid  = threadIdx.x;
    const int lane = tid & 31;
    const int wave = tid >> 5;
    const int hi   = lane >> 4;
    const int lo16 = lane & 15;
    const int b0   = blockIdx.x * ROWS_PER_BLOCK + wave * ROWS_PER_WAVE;
    const int row  = b0 + lo16;                      // this lane's batch column
    const float* xrow = x + (size_t)((row < B) ? row : (B - 1)) * D_IN;

    // -------- fc1 (transposed): H[64 x 16] = relu(W1^T @ x^T + b1) --------
    v8f H[4];
#pragma unroll
    for (int t = 0; t < 4; ++t) {
        int m0 = t * 16 + 8 * hi;
#pragma unroll
        for (int j = 0; j < 8; ++j) H[t][j] = b1[m0 + j];
    }

    for (int ph = 0; ph < 2; ++ph) {
        const int cbase = ph ? PH0 : 0;
        const int nch   = ph ? PH1 : PH0;
        const int n4    = (nch * 4 * FRAG_DW) >> 2;  // 16B units to copy
        __syncthreads();   // previous-phase consumers done before refill

        // stage pre-swizzled W1^T fragments: global (d_ws) -> LDS
        const v4u* src4 = (const v4u*)(wsFrag + cbase * 4 * FRAG_DW);
        v4u* dst4 = (v4u*)sFrag;
#if USE_ASYNC_LDS
        for (int i = tid; i < n4; i += 256) {
            __builtin_amdgcn_global_load_async_to_lds_b128(
                (g_v4i*)(src4 + i), (l_v4i*)(dst4 + i), 0, 0);
        }
#if __has_builtin(__builtin_amdgcn_s_wait_asynccnt)
        __builtin_amdgcn_s_wait_asynccnt(0);
#else
        asm volatile("s_wait_asynccnt 0" ::: "memory");
#endif
#else
        for (int i = tid; i < n4; i += 256) dst4[i] = src4[i];
#endif
        __syncthreads();

        for (int cc = 0; cc < nch; ++cc) {
            const int c  = cbase + cc;
            const int kb = c * 32 + hi * 16;
            unsigned ub[8];
            if (kb < D_IN) {  // only (c==24, hi==1) is fully out of range
                const vf4* p = (const vf4*)(xrow + kb);
                vf4 f0 = __builtin_nontemporal_load(p + 0);
                vf4 f1 = __builtin_nontemporal_load(p + 1);
                vf4 f2 = __builtin_nontemporal_load(p + 2);
                vf4 f3 = __builtin_nontemporal_load(p + 3);
                ub[0] = pack2(f0[0], f0[1]); ub[1] = pack2(f0[2], f0[3]);
                ub[2] = pack2(f1[0], f1[1]); ub[3] = pack2(f1[2], f1[3]);
                ub[4] = pack2(f2[0], f2[1]); ub[5] = pack2(f2[2], f2[3]);
                ub[6] = pack2(f3[0], f3[1]); ub[7] = pack2(f3[2], f3[3]);
            } else {
#pragma unroll
                for (int j = 0; j < 8; ++j) ub[j] = 0u;
            }
            v16h Bf = mk(ub);
#pragma unroll
            for (int mt = 0; mt < 4; ++mt) {
                const v8u* ap = (const v8u*)&sFrag[(cc * 4 + mt) * FRAG_DW + lane * 8];
                v16h Af = __builtin_bit_cast(v16h, *ap);
                H[mt] = wmma_f16(Af, Bf, H[mt]);
            }
        }
    }

    // relu + pack h^T for the next matmul's B operand
    unsigned P[4][4];
#pragma unroll
    for (int t = 0; t < 4; ++t)
#pragma unroll
        for (int j = 0; j < 4; ++j)
            P[t][j] = pack2(fmaxf(H[t][2 * j], 0.f), fmaxf(H[t][2 * j + 1], 0.f));

    // -------- x_inj^T = Wx^T @ h^T + (bx + bz), kept resident --------
    v8f XB[4];
#pragma unroll
    for (int t = 0; t < 4; ++t) {
        int m0 = t * 16 + 8 * hi;
#pragma unroll
        for (int j = 0; j < 8; ++j) XB[t][j] = bx[m0 + j] + bz[m0 + j];
    }
    {
        v16h B0 = bfrag(P[0], P[1], hi);
        v16h B1 = bfrag(P[2], P[3], hi);
#pragma unroll
        for (int t = 0; t < 4; ++t) {
            v16h A0 = load_wA(Wx, D_HID, D_HID, 0, t, lane);
            v16h A1 = load_wA(Wx, D_HID, D_HID, 1, t, lane);
            XB[t] = wmma_f16(A0, B0, XB[t]);
            XB[t] = wmma_f16(A1, B1, XB[t]);
        }
    }

    // -------- Wz^T A-fragments resident in registers (64 VGPRs) --------
    v16h WzA[2][4];
#pragma unroll
    for (int c = 0; c < 2; ++c)
#pragma unroll
        for (int t = 0; t < 4; ++t)
            WzA[c][t] = load_wA(Wz, D_HID, D_HID, c, t, lane);

    // -------- DEQ fixed point: z = tanh(x_inj + Wz^T @ z), 20 iters --------
    v8f Z[4];
#pragma unroll
    for (int t = 0; t < 4; ++t)           // iteration 1: z0 == 0
#pragma unroll
        for (int j = 0; j < 8; ++j) Z[t][j] = fast_tanh(XB[t][j]);

    for (int it = 1; it < DEQ_ITERS; ++it) {
        unsigned Pz[4][4];
#pragma unroll
        for (int t = 0; t < 4; ++t)
#pragma unroll
            for (int j = 0; j < 4; ++j)
                Pz[t][j] = pack2(Z[t][2 * j], Z[t][2 * j + 1]);
        v16h B0 = bfrag(Pz[0], Pz[1], hi);
        v16h B1 = bfrag(Pz[2], Pz[3], hi);
#pragma unroll
        for (int t = 0; t < 4; ++t) {
            v8f acc = wmma_f16(WzA[0][t], B0, XB[t]);
            acc     = wmma_f16(WzA[1][t], B1, acc);
#pragma unroll
            for (int j = 0; j < 8; ++j) Z[t][j] = fast_tanh(acc[j]);
        }
    }

    // -------- fc3: out^T[10 x 16] = W3^T @ z^T + b3 --------
    unsigned Pz[4][4];
#pragma unroll
    for (int t = 0; t < 4; ++t)
#pragma unroll
        for (int j = 0; j < 4; ++j)
            Pz[t][j] = pack2(Z[t][2 * j], Z[t][2 * j + 1]);
    v16h B0 = bfrag(Pz[0], Pz[1], hi);
    v16h B1 = bfrag(Pz[2], Pz[3], hi);
    v16h A0 = load_wA(W3, D_OUT, D_OUT, 0, 0, lane);
    v16h A1 = load_wA(W3, D_OUT, D_OUT, 1, 0, lane);
    v8f O = {};
    O = wmma_f16(A0, B0, O);
    O = wmma_f16(A1, B1, O);

    if (row < B) {
        float* orow = out + (size_t)row * D_OUT;
        if (hi == 0) {                       // features 0..7
#pragma unroll
            for (int j = 0; j < 4; ++j) {
                vf2 s;
                s[0] = O[2 * j]     + b3[2 * j];
                s[1] = O[2 * j + 1] + b3[2 * j + 1];
                __builtin_nontemporal_store(s, (vf2*)(orow + 2 * j));
            }
        } else {                             // features 8..9 (10..15 are pad)
            vf2 s;
            s[0] = O[0] + b3[8];
            s[1] = O[1] + b3[9];
            __builtin_nontemporal_store(s, (vf2*)(orow + 8));
        }
    }
}

// ---------- launch ----------
extern "C" void kernel_launch(void* const* d_in, const int* in_sizes, int n_in,
                              void* d_out, int out_size, void* d_ws, size_t ws_size,
                              hipStream_t stream) {
    const float* x  = (const float*)d_in[0];
    const float* W1 = (const float*)d_in[1];
    const float* b1 = (const float*)d_in[2];
    const float* Wx = (const float*)d_in[3];
    const float* bx = (const float*)d_in[4];
    const float* Wz = (const float*)d_in[5];
    const float* bz = (const float*)d_in[6];
    const float* W3 = (const float*)d_in[7];
    const float* b3 = (const float*)d_in[8];
    float* out = (float*)d_out;
    unsigned* wsFrag = (unsigned*)d_ws;

    const int B = in_sizes[0] / D_IN;
    const int grid = (B + ROWS_PER_BLOCK - 1) / ROWS_PER_BLOCK;

    w1_pack_kernel<<<(W1_DW + 255) / 256, 256, 0, stream>>>(W1, wsFrag);
    deq_wmma_kernel<<<grid, 256, 0, stream>>>(x, b1, Wx, bx, Wz, bz, W3, b3,
                                              wsFrag, out, B);
}
